// Attention_53841710022794
// MI455X (gfx1250) — compile-verified, matrix operand order checked by reading, not run on hardware
//
#include <hip/hip_runtime.h>

// ---------------------------------------------------------------------------
// Problem constants (from reference)
// ---------------------------------------------------------------------------
#define T_LEN 2048
#define C_DIM 4096
#define NH    32
#define NKV   8
#define HS    128
#define QK_SCALE 0.08838834764831845f   // 1/sqrt(128)

typedef __bf16 bf16;
typedef __attribute__((ext_vector_type(16))) __bf16 v16bf;
typedef __attribute__((ext_vector_type(8)))  __bf16 v8bf;
typedef __attribute__((ext_vector_type(8)))  float  v8f;
typedef __attribute__((ext_vector_type(8)))  unsigned short v8u16;

// ---------------------------------------------------------------------------
// CDNA5 async global->LDS copy (ASYNCcnt-tracked, no VGPR round-trip).
// Builtin signature (from hipcc diagnostic): param0 = int4 addrspace(1)*,
// param1 = int4 addrspace(3)*, then imm offset + cpol.
// ---------------------------------------------------------------------------
#if defined(__gfx1250__) && \
    __has_builtin(__builtin_amdgcn_global_load_async_to_lds_b128) && \
    __has_builtin(__builtin_amdgcn_s_wait_asynccnt)
#define HAVE_ASYNC_LDS 1
#else
#define HAVE_ASYNC_LDS 0
#endif

typedef int v4i __attribute__((vector_size(16)));
typedef __attribute__((address_space(1))) v4i gv4i;   // global int4
typedef __attribute__((address_space(3))) v4i lv4i;   // LDS int4

__device__ __forceinline__ void async_copy16(const unsigned short* g,
                                             unsigned short* l) {
#if HAVE_ASYNC_LDS
    __builtin_amdgcn_global_load_async_to_lds_b128((gv4i*)g, (lv4i*)l, 0, 0);
#else
    *(v8u16*)l = *(const v8u16*)g;
#endif
}

__device__ __forceinline__ void async_wait() {
#if HAVE_ASYNC_LDS
    __builtin_amdgcn_s_wait_asynccnt(0);
#endif
}

// float -> bf16 (round-to-nearest-even)
__device__ __forceinline__ unsigned short f2bf(float f) {
    unsigned int u = __float_as_uint(f);
    u += 0x7FFFu + ((u >> 16) & 1u);
    return (unsigned short)(u >> 16);
}

// WMMA wrapper: D = A(16x32 bf16) * B(32x16 bf16) + C(16x16 f32)
__device__ __forceinline__ v8f wmma_bf16(v16bf a, v16bf b, v8f c) {
    return __builtin_amdgcn_wmma_f32_16x16x32_bf16(
        /*neg_a=*/false, a, /*neg_b=*/false, b,
        /*c_mod=*/(short)0, c, /*reuse_a=*/false, /*reuse_b=*/false);
}

// Load one 16x32 bf16 operand tile (A layout; B layout mirrors it and reads
// identically from a K-contiguous source). Lane L (row = L&15) holds K chunks
// [h*8 .. h*8+7] and [16+h*8 .. 16+h*8+7] where h = L>>4. `ld` is the row
// stride in elements (must be a multiple of 8 for 16B alignment).
__device__ __forceinline__ v16bf load_op(const unsigned short* p, int ld) {
    int lane = threadIdx.x & 31;
    int r = lane & 15, h = lane >> 4;
    const bf16* q = (const bf16*)(p + r * ld + h * 8);
    v8bf lo = *(const v8bf*)(q);
    v8bf hi = *(const v8bf*)(q + 16);
    return __builtin_shufflevector(lo, hi, 0,1,2,3,4,5,6,7,8,9,10,11,12,13,14,15);
}

// Reductions across the 16 lanes of each wave half (xor masks 1,2,4,8 stay
// inside a half — matches the C-tile row ownership).
__device__ __forceinline__ float redmax16(float v) {
    #pragma unroll
    for (int m = 1; m < 16; m <<= 1) v = fmaxf(v, __shfl_xor(v, m, 32));
    return v;
}
__device__ __forceinline__ float redsum16(float v) {
    #pragma unroll
    for (int m = 1; m < 16; m <<= 1) v += __shfl_xor(v, m, 32);
    return v;
}

// ---------------------------------------------------------------------------
// Kernel 1: f32 -> bf16 conversion
// ---------------------------------------------------------------------------
__global__ void cvt_bf16_kernel(const float* __restrict__ src,
                                unsigned short* __restrict__ dst, int n) {
    int i = blockIdx.x * blockDim.x + threadIdx.x;
    if (i < n) dst[i] = f2bf(src[i]);
}

// ---------------------------------------------------------------------------
// Kernel 2: GEMM  out[m,n] = sum_k A[m,k]*W[n,k] + bias[n]
// A: MxK bf16 row-major, W: NxK bf16 row-major, out: MxN f32.
// Block = 256 threads = 8 waves (4x2), block tile 128x128, wave tile 32x64.
// Double-buffered LDS fed by async global->LDS copies: while WMMA consumes
// buffer `cur`, the next K-slice streams into buffer `cur^1`.
// ---------------------------------------------------------------------------
__global__ void __launch_bounds__(256)
gemm_bf16_kernel(const unsigned short* __restrict__ A,
                 const unsigned short* __restrict__ W,
                 const float* __restrict__ bias,
                 float* __restrict__ out,
                 int M, int N, int K) {
    __shared__ unsigned short Alds[2][128 * 32];
    __shared__ unsigned short Wlds[2][128 * 32];

    const int tid  = threadIdx.x;
    const int lane = tid & 31, wave = tid >> 5;
    const int wm = wave >> 1, wn = wave & 1;          // 4 x 2 wave grid
    const int m0 = blockIdx.y * 128, n0 = blockIdx.x * 128;
    const int r16 = lane & 15, half = lane >> 4;

    // Stage the 128x32 A and W tiles at K-offset kk into LDS buffer `buf`.
    auto stage = [&](int buf, int kk) {
        #pragma unroll
        for (int i = 0; i < 2; i++) {
            int c = tid * 2 + i;                      // 0..511 16B chunks
            int row = c >> 2, col8 = (c & 3) * 8;
            async_copy16(&A[(size_t)(m0 + row) * K + kk + col8],
                         &Alds[buf][row * 32 + col8]);
            async_copy16(&W[(size_t)(n0 + row) * K + kk + col8],
                         &Wlds[buf][row * 32 + col8]);
        }
    };

    v8f acc[2][4];
    #pragma unroll
    for (int mt = 0; mt < 2; mt++)
        #pragma unroll
        for (int nt = 0; nt < 4; nt++) acc[mt][nt] = {};

    stage(0, 0);                                      // prologue
    for (int kk = 0; kk < K; kk += 32) {
        const int cur = (kk >> 5) & 1;
        async_wait();                                 // own async copies done
        __syncthreads();                              // everyone's copies done
        if (kk + 32 < K) {
            stage(cur ^ 1, kk + 32);                  // overlap next copy
            // warm L2 for the tile after next (beyond the double buffer)
            if (kk + 64 < K) {
                __builtin_prefetch(&A[(size_t)(m0 + (tid >> 1)) * K + kk + 64], 0, 3);
                __builtin_prefetch(&W[(size_t)(n0 + (tid >> 1)) * K + kk + 64], 0, 3);
            }
        }

        v16bf a[2], b[4];
        #pragma unroll
        for (int mt = 0; mt < 2; mt++)
            a[mt] = load_op(&Alds[cur][(wm * 32 + mt * 16) * 32], 32);
        #pragma unroll
        for (int nt = 0; nt < 4; nt++)
            b[nt] = load_op(&Wlds[cur][(wn * 64 + nt * 16) * 32], 32);
        #pragma unroll
        for (int mt = 0; mt < 2; mt++)
            #pragma unroll
            for (int nt = 0; nt < 4; nt++)
                acc[mt][nt] = wmma_bf16(a[mt], b[nt], acc[mt][nt]);
    }

    #pragma unroll
    for (int mt = 0; mt < 2; mt++)
        #pragma unroll
        for (int nt = 0; nt < 4; nt++)
            #pragma unroll
            for (int e = 0; e < 8; e++) {
                int row = m0 + wm * 32 + mt * 16 + e + 8 * half;
                int col = n0 + wn * 64 + nt * 16 + r16;
                out[(size_t)row * N + col] = acc[mt][nt][e] + bias[col];
            }
}

// ---------------------------------------------------------------------------
// Kernel 3: RoPE + reorder (T, H*HS) f32 -> (H, T, HS) bf16
// ---------------------------------------------------------------------------
__global__ void rope_reorder_kernel(const float* __restrict__ src,
                                    unsigned short* __restrict__ dst,
                                    int nheads, int do_rope) {
    int idx = blockIdx.x * blockDim.x + threadIdx.x;
    int total = T_LEN * nheads * (HS / 2);
    if (idx >= total) return;
    int i = idx & 63;                 // HS/2 = 64
    int h = (idx >> 6) % nheads;
    int t = idx / (nheads * 64);

    float a = src[((size_t)t * nheads + h) * HS + i];
    float b = src[((size_t)t * nheads + h) * HS + i + 64];
    float o1 = a, o2 = b;
    if (do_rope) {
        // inv_freq = 10000^(-2i/128)
        float ang = (float)t * __expf((-2.0f * (float)i / 128.0f) * 9.210340372f);
        float c = cosf(ang), s = sinf(ang);
        o1 = a * c - b * s;           // x1*cos + (-x2)*sin
        o2 = b * c + a * s;           // x2*cos + ( x1)*sin
    }
    dst[((size_t)h * T_LEN + t) * HS + i]      = f2bf(o1);
    dst[((size_t)h * T_LEN + t) * HS + i + 64] = f2bf(o2);
}

// ---------------------------------------------------------------------------
// Kernel 4: causal flash attention with WMMA.
// grid = (T/64 query blocks, NH heads), block = 128 threads = 4 waves.
// Each wave owns 16 query rows; key blocks of 32 staged in LDS (K via async
// global->LDS DMA, V transposed through VGPRs).
// Output: ybf (T, NH*HS) bf16.
// ---------------------------------------------------------------------------
__global__ void __launch_bounds__(128)
attn_kernel(const unsigned short* __restrict__ qbf,
            const unsigned short* __restrict__ kbf,
            const unsigned short* __restrict__ vbf,
            unsigned short* __restrict__ ybf) {
    __shared__ unsigned short Klds[32 * HS];        // key block, row-major
    __shared__ unsigned short Vt[HS * 32];          // value block, transposed
    __shared__ unsigned short Plds[4][16 * 32];     // per-wave P tile

    const int qblk = blockIdx.x, h = blockIdx.y;
    const int kvh  = h >> 2;                        // GROUPS = 4
    const int tid  = threadIdx.x, lane = tid & 31, wave = tid >> 5;
    const int r16  = lane & 15, half = lane >> 4;
    const int qrow0 = qblk * 64 + wave * 16;

    // Q operand tiles: 16 rows x 128 (4 chunks of K=32), straight from global.
    const unsigned short* qptr = qbf + ((size_t)h * T_LEN + qrow0) * HS;
    v16bf qop[4];
    #pragma unroll
    for (int c = 0; c < 4; c++) qop[c] = load_op(qptr + c * 32, HS);

    v8f Oacc[8];
    float m_i[8], l_i[8];
    #pragma unroll
    for (int dt = 0; dt < 8; dt++) Oacc[dt] = {};
    #pragma unroll
    for (int e = 0; e < 8; e++) { m_i[e] = -1e30f; l_i[e] = 0.0f; }

    const int nkb = qblk * 2 + 2;                   // causal: keys <= qblk end
    for (int kbi = 0; kbi < nkb; kbi++) {
        const int kb = kbi * 32;
        __syncthreads();                            // prior compute finished
        // Stage K (async DMA, row-major) and V (VGPR transpose) blocks.
        for (int c = tid; c < 512; c += 128) {      // 512 16B chunks
            int row = c >> 4, col8 = (c & 15) * 8;
            size_t gbase = ((size_t)kvh * T_LEN + kb + row) * HS + col8;
            async_copy16(&kbf[gbase], &Klds[row * HS + col8]);
            v8u16 vv = *(const v8u16*)&vbf[gbase];
            #pragma unroll
            for (int j = 0; j < 8; j++) Vt[(col8 + j) * 32 + row] = vv[j];
        }
        async_wait();
        __syncthreads();

        // S (16x32) = Q (16x128) @ K^T, two 16x16 C tiles.
        v8f S0 = {}, S1 = {};
        #pragma unroll
        for (int c = 0; c < 4; c++) {
            v16bf b0 = load_op(&Klds[0 * HS + c * 32], HS);
            v16bf b1 = load_op(&Klds[16 * HS + c * 32], HS);
            S0 = wmma_bf16(qop[c], b0, S0);
            S1 = wmma_bf16(qop[c], b1, S1);
        }

        // Online softmax update (row r = e + 8*half, col = lane&15 [+16]).
        #pragma unroll
        for (int e = 0; e < 8; e++) {
            int qrow = qrow0 + e + 8 * half;
            int c0 = kb + r16;
            float s0 = S0[e] * QK_SCALE;
            float s1 = S1[e] * QK_SCALE;
            if (c0 > qrow)      s0 = -1e30f;
            if (c0 + 16 > qrow) s1 = -1e30f;
            float rm   = redmax16(fmaxf(s0, s1));
            float mnew = fmaxf(m_i[e], rm);
            float sc   = __expf(m_i[e] - mnew);
            float p0   = __expf(s0 - mnew);
            float p1   = __expf(s1 - mnew);
            float rsum = redsum16(p0 + p1);
            l_i[e] = l_i[e] * sc + rsum;
            m_i[e] = mnew;
            #pragma unroll
            for (int dt = 0; dt < 8; dt++) Oacc[dt][e] *= sc;
            Plds[wave][(e + 8 * half) * 32 + r16]      = f2bf(p0);
            Plds[wave][(e + 8 * half) * 32 + 16 + r16] = f2bf(p1);
        }

        // O (16x128) += P (16x32) @ V (32x128); B operand from transposed Vt.
        v16bf pa = load_op(&Plds[wave][0], 32);
        #pragma unroll
        for (int dt = 0; dt < 8; dt++) {
            v16bf bv = load_op(&Vt[dt * 16 * 32], 32);
            Oacc[dt] = wmma_bf16(pa, bv, Oacc[dt]);
        }
    }

    // Normalize and store to (T, NH*HS) bf16.
    #pragma unroll
    for (int e = 0; e < 8; e++) {
        int qrow = qrow0 + e + 8 * half;
        float inv = 1.0f / l_i[e];
        #pragma unroll
        for (int dt = 0; dt < 8; dt++)
            ybf[(size_t)qrow * (NH * HS) + h * HS + dt * 16 + r16] =
                f2bf(Oacc[dt][e] * inv);
    }
}

// ---------------------------------------------------------------------------
// Host launcher
// ---------------------------------------------------------------------------
extern "C" void kernel_launch(void* const* d_in, const int* in_sizes, int n_in,
                              void* d_out, int out_size, void* d_ws, size_t ws_size,
                              hipStream_t stream) {
    const float* x  = (const float*)d_in[0];
    const float* Wq = (const float*)d_in[1];
    const float* bq = (const float*)d_in[2];
    const float* Wk = (const float*)d_in[3];
    const float* bk = (const float*)d_in[4];
    const float* Wv = (const float*)d_in[5];
    const float* bv = (const float*)d_in[6];
    const float* Wc = (const float*)d_in[7];
    const float* bc = (const float*)d_in[8];
    float* out = (float*)d_out;

    const size_t MB = 1ull << 20;
    char* ws = (char*)d_ws;
    unsigned short* xb  = (unsigned short*)(ws +   0 * MB); // 16 MB
    unsigned short* wqb = (unsigned short*)(ws +  16 * MB); // 32 MB
    unsigned short* wkb = (unsigned short*)(ws +  48 * MB); //  8 MB
    unsigned short* wvb = (unsigned short*)(ws +  56 * MB); //  8 MB
    unsigned short* wcb = (unsigned short*)(ws +  64 * MB); // 32 MB
    float*          qf  = (float*)         (ws +  96 * MB); // 32 MB
    float*          kf  = (float*)         (ws + 128 * MB); //  8 MB
    float*          vf  = (float*)         (ws + 136 * MB); //  8 MB
    unsigned short* qbf = (unsigned short*)(ws + 144 * MB); // 16 MB
    unsigned short* kbf = (unsigned short*)(ws + 160 * MB); //  4 MB
    unsigned short* vbf = (unsigned short*)(ws + 164 * MB); //  4 MB
    unsigned short* ybf = (unsigned short*)(ws + 168 * MB); // 16 MB

    // 1) f32 -> bf16 conversions.
    auto cvt = [&](const float* s, unsigned short* d, int n) {
        cvt_bf16_kernel<<<(n + 255) / 256, 256, 0, stream>>>(s, d, n);
    };
    cvt(x,  xb,  T_LEN * C_DIM);
    cvt(Wq, wqb, (NH  * HS) * C_DIM);
    cvt(Wk, wkb, (NKV * HS) * C_DIM);
    cvt(Wv, wvb, (NKV * HS) * C_DIM);
    cvt(Wc, wcb, C_DIM * (NH * HS));

    // 2) Q/K/V projections (f32 out + bias).
    gemm_bf16_kernel<<<dim3((NH * HS) / 128, T_LEN / 128), 256, 0, stream>>>(
        xb, wqb, bq, qf, T_LEN, NH * HS, C_DIM);
    gemm_bf16_kernel<<<dim3((NKV * HS) / 128, T_LEN / 128), 256, 0, stream>>>(
        xb, wkb, bk, kf, T_LEN, NKV * HS, C_DIM);
    gemm_bf16_kernel<<<dim3((NKV * HS) / 128, T_LEN / 128), 256, 0, stream>>>(
        xb, wvb, bv, vf, T_LEN, NKV * HS, C_DIM);

    // 3) RoPE + head-major reorder to bf16.
    {
        int nq = T_LEN * NH * (HS / 2);
        int nk = T_LEN * NKV * (HS / 2);
        rope_reorder_kernel<<<(nq + 255) / 256, 256, 0, stream>>>(qf, qbf, NH, 1);
        rope_reorder_kernel<<<(nk + 255) / 256, 256, 0, stream>>>(kf, kbf, NKV, 1);
        rope_reorder_kernel<<<(nk + 255) / 256, 256, 0, stream>>>(vf, vbf, NKV, 0);
    }

    // 4) Causal GQA flash attention.
    attn_kernel<<<dim3(T_LEN / 64, NH), 128, 0, stream>>>(qbf, kbf, vbf, ybf);

    // 5) Output projection -> d_out (f32).
    gemm_bf16_kernel<<<dim3(C_DIM / 128, T_LEN / 128), 256, 0, stream>>>(
        ybf, wcb, bc, out, T_LEN, C_DIM, NH * HS);
}
